// TransformerBlock_82437602280001
// MI455X (gfx1250) — matrix operand order repaired and checked by reading, not compiled, not run on hardware
//
#include <hip/hip_runtime.h>
#include <cstdint>

#define C_CH  128
#define H_DIM 64
#define W_DIM 256
#define HW    16384         // H_DIM*W_DIM
#define B_DIM 4
#define P_TOT 65536         // B_DIM*HW
#define NHEAD 4
#define HEADD 32
#define E_CH  256

typedef __attribute__((ext_vector_type(16))) _Float16 v16h;
typedef __attribute__((ext_vector_type(8)))  float    v8f;
typedef __attribute__((ext_vector_type(4)))  float    f32x4;

// ---------------------------------------------------------------- WMMA helpers
__device__ inline v8f wmma_f16(v16h a, v16h b, v8f c) {
  return __builtin_amdgcn_wmma_f32_16x16x32_f16(false, a, false, b, (short)0, c,
                                                false, false);
}

// A-matrix 16x32 f16 fragment from row-major A[m][k] (ld halves).
// lane<16: row=lane, K 0..7 & 16..23 ; lane>=16: row=lane-16, K 8..15 & 24..31
__device__ inline v16h frag_a16x32(const _Float16* base, int ld, int lane) {
  int row = lane & 15;
  int kb  = (lane >> 4) << 3;                 // 0 or 8
  const _Float16* p = base + (size_t)row * ld + kb;
  union { v16h v; f32x4 q[2]; } u;
  u.q[0] = *reinterpret_cast<const f32x4*>(p);
  u.q[1] = *reinterpret_cast<const f32x4*>(p + 16);
  return u.v;
}

// B-matrix 32x16 f16 fragment from transposed storage Bt[n][k] (ld halves).
// lane<16: col=lane, K 0..7 & 16..23 ; lane>=16: col=lane-16, K 8..15 & 24..31
__device__ inline v16h frag_b32x16(const _Float16* base, int ld, int lane) {
  int col = lane & 15;
  int kb  = (lane >> 4) << 3;                 // 0 or 8
  const _Float16* p = base + (size_t)col * ld + kb;
  union { v16h v; f32x4 q[2]; } u;
  u.q[0] = *reinterpret_cast<const f32x4*>(p);
  u.q[1] = *reinterpret_cast<const f32x4*>(p + 16);
  return u.v;
}

// ---------------------------------------------------------------- LayerNorm #1
// cat1 = [f1;f2], cat2 = [f2;f1]; LN over channels; write pixel-major f16.
__global__ __launch_bounds__(256) void ln_in_kernel(
    const float* __restrict__ f1, const float* __restrict__ f2,
    const float* __restrict__ g1, const float* __restrict__ b1,
    const float* __restrict__ g2, const float* __restrict__ b2,
    _Float16* __restrict__ x1h, _Float16* __restrict__ x2h) {
  int p  = blockIdx.x * 256 + threadIdx.x;
  int b  = p >> 14, hw = p & (HW - 1);
  const float* s1 = (b < 2) ? f1 + (size_t)b * C_CH * HW
                            : f2 + (size_t)(b - 2) * C_CH * HW;
  const float* s2 = (b < 2) ? f2 + (size_t)b * C_CH * HW
                            : f1 + (size_t)(b - 2) * C_CH * HW;
  {
    float s = 0.f, sq = 0.f;
    for (int c = 0; c < C_CH; ++c) { float x = s1[(size_t)c * HW + hw]; s += x; sq += x * x; }
    float m = s * (1.f / C_CH), v = sq * (1.f / C_CH) - m * m;
    float inv = rsqrtf(v + 1e-6f);
    for (int c = 0; c < C_CH; ++c) {
      float x = s1[(size_t)c * HW + hw];
      x1h[(size_t)p * C_CH + c] = (_Float16)(((x - m) * inv) * g1[c] + b1[c]);
    }
  }
  {
    float s = 0.f, sq = 0.f;
    for (int c = 0; c < C_CH; ++c) { float x = s2[(size_t)c * HW + hw]; s += x; sq += x * x; }
    float m = s * (1.f / C_CH), v = sq * (1.f / C_CH) - m * m;
    float inv = rsqrtf(v + 1e-6f);
    for (int c = 0; c < C_CH; ++c) {
      float x = s2[(size_t)c * HW + hw];
      x2h[(size_t)p * C_CH + c] = (_Float16)(((x - m) * inv) * g2[c] + b2[c]);
    }
  }
}

// ---------------------------------------------------------------- LayerNorm #2
__global__ __launch_bounds__(256) void ln_mid_kernel(
    const float* __restrict__ catm, const float* __restrict__ g,
    const float* __restrict__ bb, _Float16* __restrict__ xh) {
  int p = blockIdx.x * 256 + threadIdx.x;
  int b = p >> 14, hw = p & (HW - 1);
  const float* src = catm + (size_t)b * C_CH * HW;
  float s = 0.f, sq = 0.f;
  for (int c = 0; c < C_CH; ++c) { float x = src[(size_t)c * HW + hw]; s += x; sq += x * x; }
  float m = s * (1.f / C_CH), v = sq * (1.f / C_CH) - m * m;
  float inv = rsqrtf(v + 1e-5f);
  for (int c = 0; c < C_CH; ++c) {
    float x = src[(size_t)c * HW + hw];
    xh[(size_t)p * C_CH + c] = (_Float16)(((x - m) * inv) * g[c] + bb[c]);
  }
}

// ---------------------------------------------------------------- generic GEMM
// Out[p][n] = epilogue( sum_k X[p][k] * Wt[n][k] + bias[n] )
// mode 0: f16 store    1: SiLU,f16   2: SiLU + resid_h, f16
// mode 3: f32 out[(b,C,h,w)] = cat1_src + acc   4: f32 out += acc
__global__ __launch_bounds__(256) void gemm_kernel(
    const _Float16* __restrict__ X, int ldx,
    const float* __restrict__ Wt, const float* __restrict__ bias,
    int K, int mode,
    const _Float16* __restrict__ residh,
    const float* __restrict__ f1, const float* __restrict__ f2,
    _Float16* __restrict__ outh, int ldo,
    float* __restrict__ outf) {
  __shared__ _Float16 sA[128 * 32];
  __shared__ _Float16 sB[64 * 32];
  int tid = threadIdx.x, lane = tid & 31, wid = tid >> 5;
  int wm = wid & 3, wn = wid >> 2;           // 4 waves along M, 2 along N
  int pblk = blockIdx.y * 128, nblk = blockIdx.x * 64;
  const v8f vz = {0, 0, 0, 0, 0, 0, 0, 0};
  v8f acc[2][2] = {{vz, vz}, {vz, vz}};

  for (int k0 = 0; k0 < K; k0 += 32) {
    {                                        // stage A: 128x32 f16
      int r = tid >> 1, ch = (tid & 1) << 4;
      const _Float16* src = X + (size_t)(pblk + r) * ldx + k0 + ch;
      f32x4* dst = reinterpret_cast<f32x4*>(&sA[r * 32 + ch]);
      dst[0] = *reinterpret_cast<const f32x4*>(src);
      dst[1] = *reinterpret_cast<const f32x4*>(src + 8);
    }
    {                                        // stage B: 64x32, f32 -> f16
      int r = tid >> 2, off = (tid & 3) << 3;
      const float* src = Wt + (size_t)(nblk + r) * K + k0 + off;
      f32x4 w0 = *reinterpret_cast<const f32x4*>(src);
      f32x4 w1 = *reinterpret_cast<const f32x4*>(src + 4);
      _Float16* dst = &sB[r * 32 + off];
#pragma unroll
      for (int i = 0; i < 4; ++i) { dst[i] = (_Float16)w0[i]; dst[4 + i] = (_Float16)w1[i]; }
    }
    __syncthreads();
    v16h a0 = frag_a16x32(&sA[(wm * 32 + 0)  * 32], 32, lane);
    v16h a1 = frag_a16x32(&sA[(wm * 32 + 16) * 32], 32, lane);
    v16h b0 = frag_b32x16(&sB[(wn * 32 + 0)  * 32], 32, lane);
    v16h b1 = frag_b32x16(&sB[(wn * 32 + 16) * 32], 32, lane);
    acc[0][0] = wmma_f16(a0, b0, acc[0][0]);
    acc[0][1] = wmma_f16(a0, b1, acc[0][1]);
    acc[1][0] = wmma_f16(a1, b0, acc[1][0]);
    acc[1][1] = wmma_f16(a1, b1, acc[1][1]);
    __syncthreads();
  }

  int mg = (lane >> 4) << 3, nl = lane & 15;
#pragma unroll
  for (int mt = 0; mt < 2; ++mt)
#pragma unroll
    for (int nt = 0; nt < 2; ++nt) {
      int ncol = nblk + wn * 32 + nt * 16 + nl;
      float bv = bias[ncol];
#pragma unroll
      for (int r = 0; r < 8; ++r) {
        int prow = pblk + wm * 32 + mt * 16 + mg + r;
        float v = acc[mt][nt][r] + bv;
        if (mode == 0) {
          outh[(size_t)prow * ldo + ncol] = (_Float16)v;
        } else if (mode == 1) {
          v = v / (1.f + __expf(-v));
          outh[(size_t)prow * ldo + ncol] = (_Float16)v;
        } else if (mode == 2) {
          v = v / (1.f + __expf(-v));
          v += (float)residh[(size_t)prow * ldo + ncol];
          outh[(size_t)prow * ldo + ncol] = (_Float16)v;
        } else if (mode == 3) {
          int b = prow >> 14, hw = prow & (HW - 1);
          float src = (b < 2) ? f1[((size_t)b * C_CH + ncol) * HW + hw]
                              : f2[((size_t)(b - 2) * C_CH + ncol) * HW + hw];
          outf[((size_t)b * C_CH + ncol) * HW + hw] = src + v;
        } else {
          int b = prow >> 14, hw = prow & (HW - 1);
          outf[((size_t)b * C_CH + ncol) * HW + hw] += v;
        }
      }
    }
}

// ---------------------------------------------------------------- attention
// One workgroup per (b, head, row h): Q,K,V are 32x256; 2-pass softmax over W.
__global__ __launch_bounds__(256) void attn_kernel(
    const _Float16* __restrict__ qh, const _Float16* __restrict__ kvh,
    _Float16* __restrict__ oh) {
  __shared__ _Float16 Vd[32][264];           // V as [d][j], contiguous in j
  __shared__ float    Vsum[32];
  __shared__ _Float16 Pt[8][16 * 32];        // per-wave score tile bounce
  int bid = blockIdx.x;
  int b = bid >> 8, nh = (bid >> 6) & 3, h = bid & 63;
  int pixbase = b * HW + h * W_DIM;
  int cq = nh * HEADD;
  int tid = threadIdx.x, lane = tid & 31, wid = tid >> 5;

  {                                          // stage V: thread t = column j
    const _Float16* src = kvh + (size_t)(pixbase + tid) * E_CH + C_CH + cq;
#pragma unroll
    for (int c4 = 0; c4 < 4; ++c4) {
      union { f32x4 q; _Float16 e[8]; } u;
      u.q = *reinterpret_cast<const f32x4*>(src + c4 * 8);
#pragma unroll
      for (int i = 0; i < 8; ++i) Vd[c4 * 8 + i][tid] = u.e[i];
    }
  }
  __syncthreads();
  if (tid < 32) {
    float s = 0.f;
    for (int j = 0; j < W_DIM; ++j) s += (float)Vd[tid][j];
    Vsum[tid] = s;
  }
  __syncthreads();

  const float scale  = 0.17677669529663687f; // 1/sqrt(32)
  const float eps    = 1e-6f;
  const float renorm = 1.f / (1.f + (float)W_DIM * eps);
  const v8f vz = {0, 0, 0, 0, 0, 0, 0, 0};
  int mg = (lane >> 4) << 3, nl = lane & 15;

  for (int strip = wid; strip < 16; strip += 8) {
    int i0 = strip * 16;
    v16h qa = frag_a16x32(qh + (size_t)(pixbase + i0) * C_CH + cq, C_CH, lane);

    // ---- pass 1: row max
    v8f mx;
#pragma unroll
    for (int r = 0; r < 8; ++r) mx[r] = -1e30f;
    for (int nj = 0; nj < 16; ++nj) {
      v16h kb = frag_b32x16(kvh + (size_t)(pixbase + nj * 16) * E_CH + cq, E_CH, lane);
      v8f s = wmma_f16(qa, kb, vz);
#pragma unroll
      for (int r = 0; r < 8; ++r) mx[r] = fmaxf(mx[r], s[r]);
    }
#pragma unroll
    for (int m = 1; m <= 8; m <<= 1)
#pragma unroll
      for (int r = 0; r < 8; ++r) mx[r] = fmaxf(mx[r], __shfl_xor(mx[r], m, 32));

    // ---- pass 2: exp, row sum, O += P @ V^T
    v8f ls = vz, oacc0 = vz, oacc1 = vz;
    _Float16* pt = Pt[wid];
    for (int njp = 0; njp < 8; ++njp) {
      v16h kb0 = frag_b32x16(kvh + (size_t)(pixbase + njp * 32)      * E_CH + cq, E_CH, lane);
      v16h kb1 = frag_b32x16(kvh + (size_t)(pixbase + njp * 32 + 16) * E_CH + cq, E_CH, lane);
      v8f s0 = wmma_f16(qa, kb0, vz);
      v8f s1 = wmma_f16(qa, kb1, vz);
#pragma unroll
      for (int r = 0; r < 8; ++r) {
        float e0 = __expf((s0[r] - mx[r]) * scale);
        float e1 = __expf((s1[r] - mx[r]) * scale);
        ls[r] += e0 + e1;
        pt[(mg + r) * 32 + nl]      = (_Float16)e0;
        pt[(mg + r) * 32 + 16 + nl] = (_Float16)e1;
      }
      asm volatile("s_wait_dscnt 0" ::: "memory");  // same-wave LDS RAW
      v16h pa  = frag_a16x32(pt, 32, lane);
      v16h vb0 = frag_b32x16(&Vd[0][njp * 32],  264, lane);
      v16h vb1 = frag_b32x16(&Vd[16][njp * 32], 264, lane);
      oacc0 = wmma_f16(pa, vb0, oacc0);
      oacc1 = wmma_f16(pa, vb1, oacc1);
    }
#pragma unroll
    for (int m = 1; m <= 8; m <<= 1)
#pragma unroll
      for (int r = 0; r < 8; ++r) ls[r] += __shfl_xor(ls[r], m, 32);

    // ---- epilogue: (P/l + eps*1*V) / (1 + W*eps)
#pragma unroll
    for (int r = 0; r < 8; ++r) {
      int row = i0 + mg + r;
      float inv = 1.f / ls[r];
      float o0 = (oacc0[r] * inv + eps * Vsum[nl])      * renorm;
      float o1 = (oacc1[r] * inv + eps * Vsum[16 + nl]) * renorm;
      oh[(size_t)(pixbase + row) * C_CH + cq + nl]      = (_Float16)o0;
      oh[(size_t)(pixbase + row) * C_CH + cq + 16 + nl] = (_Float16)o1;
    }
  }
}

// ---------------------------------------------------------------- depthwise 3/5/7
__global__ __launch_bounds__(256) void dw_kernel(
    const _Float16* __restrict__ y,
    const float* __restrict__ w3, const float* __restrict__ b3,
    const float* __restrict__ w5, const float* __restrict__ b5,
    const float* __restrict__ w7, const float* __restrict__ b7,
    _Float16* __restrict__ dwcat) {
  __shared__ float sw3[8][9], sw5[8][25], sw7[8][49];
  int cg = blockIdx.y * 8;
  int tid = threadIdx.x;
  for (int i = tid; i < 72;  i += 256) sw3[i / 9][i % 9]   = w3[(size_t)(cg + i / 9)  * 9  + i % 9];
  for (int i = tid; i < 200; i += 256) sw5[i / 25][i % 25] = w5[(size_t)(cg + i / 25) * 25 + i % 25];
  for (int i = tid; i < 392; i += 256) sw7[i / 49][i % 49] = w7[(size_t)(cg + i / 49) * 49 + i % 49];
  __syncthreads();
  int p = blockIdx.x * 256 + tid;
  int b = p >> 14, hw = p & (HW - 1), h = hw >> 8, w = hw & 255;
  float a3[8] = {}, a5[8] = {}, a7[8] = {};
  for (int dy = -3; dy <= 3; ++dy) {
    int h2 = h + dy;
    if (h2 < 0 || h2 >= H_DIM) continue;
    for (int dx = -3; dx <= 3; ++dx) {
      int w2 = w + dx;
      if (w2 < 0 || w2 >= W_DIM) continue;
      const _Float16* src = y + (size_t)((b << 14) | (h2 << 8) | w2) * E_CH + cg;
      union { f32x4 q; _Float16 e[8]; } u;
      u.q = *reinterpret_cast<const f32x4*>(src);
      int  t7  = (dy + 3) * 7 + (dx + 3);
      bool in5 = (dy >= -2 && dy <= 2 && dx >= -2 && dx <= 2);
      bool in3 = (dy >= -1 && dy <= 1 && dx >= -1 && dx <= 1);
      int  t5  = (dy + 2) * 5 + (dx + 2);
      int  t3  = (dy + 1) * 3 + (dx + 1);
#pragma unroll
      for (int i = 0; i < 8; ++i) {
        float x = (float)u.e[i];
        a7[i] += x * sw7[i][t7];
        if (in5) a5[i] += x * sw5[i][t5];
        if (in3) a3[i] += x * sw3[i][t3];
      }
    }
  }
  _Float16* d = dwcat + (size_t)p * 768;
#pragma unroll
  for (int i = 0; i < 8; ++i) {
    d[cg + i]       = (_Float16)(a3[i] + b3[cg + i]);
    d[256 + cg + i] = (_Float16)(a5[i] + b5[cg + i]);
    d[512 + cg + i] = (_Float16)(a7[i] + b7[cg + i]);
  }
}

// ---------------------------------------------------------------- launch
extern "C" void kernel_launch(void* const* d_in, const int* in_sizes, int n_in,
                              void* d_out, int out_size, void* d_ws, size_t ws_size,
                              hipStream_t stream) {
  (void)in_sizes; (void)n_in; (void)out_size; (void)ws_size;
  const float* feats1 = (const float*)d_in[0];
  const float* feats2 = (const float*)d_in[1];
  const float* an_g   = (const float*)d_in[2];
  const float* an_b   = (const float*)d_in[3];
  const float* anc_g  = (const float*)d_in[4];
  const float* anc_b  = (const float*)d_in[5];
  const float* wq     = (const float*)d_in[6];
  const float* bq     = (const float*)d_in[7];
  const float* wkv    = (const float*)d_in[8];
  const float* bkv    = (const float*)d_in[9];
  const float* wo     = (const float*)d_in[10];
  const float* bo     = (const float*)d_in[11];
  const float* fn_g   = (const float*)d_in[12];
  const float* fn_b   = (const float*)d_in[13];
  const float* w_in   = (const float*)d_in[14];
  const float* b_in   = (const float*)d_in[15];
  const float* w_dw3  = (const float*)d_in[16];
  const float* b_dw3  = (const float*)d_in[17];
  const float* w_dw5  = (const float*)d_in[18];
  const float* b_dw5  = (const float*)d_in[19];
  const float* w_dw7  = (const float*)d_in[20];
  const float* b_dw7  = (const float*)d_in[21];
  const float* w_pw   = (const float*)d_in[22];
  const float* b_pw   = (const float*)d_in[23];
  const float* w_out  = (const float*)d_in[24];
  const float* b_out  = (const float*)d_in[25];
  float* outf = (float*)d_out;

  // workspace layout (lifetime-overlapped), total 160 MB
  char* ws = (char*)d_ws;
  const size_t MB = 1u << 20;
  _Float16* x1h = (_Float16*)(ws + 0);        // [0,16M)
  _Float16* x2h = (_Float16*)(ws + 16 * MB);  // [16,32M)
  _Float16* oat = (_Float16*)(ws + 0);        // over x1h (x1h dead)
  _Float16* xyh = (_Float16*)(ws + 16 * MB);  // over x2h (x2h dead)
  _Float16* zh  = (_Float16*)(ws + 0);        // [0,32M), over oat/xyh (dead)
  _Float16* qh  = (_Float16*)(ws + 32 * MB);  // [32,48M)
  _Float16* yh  = (_Float16*)(ws + 32 * MB);  // [32,64M), over qh (dead)
  _Float16* kvh = (_Float16*)(ws + 64 * MB);  // [64,96M)
  _Float16* dwc = (_Float16*)(ws + 64 * MB);  // [64,160M), over kvh (dead)

  dim3 blk(256);
  // 1) dual channel-LN -> pixel-major f16
  ln_in_kernel<<<P_TOT / 256, blk, 0, stream>>>(feats1, feats2, an_g, an_b,
                                                anc_g, anc_b, x1h, x2h);
  // 2) q = x1 @ wq^T + bq
  gemm_kernel<<<dim3(2, P_TOT / 128), blk, 0, stream>>>(
      x1h, 128, wq, bq, 128, 0, nullptr, nullptr, nullptr, qh, 128, nullptr);
  // 3) kv = x2 @ wkv^T + bkv
  gemm_kernel<<<dim3(4, P_TOT / 128), blk, 0, stream>>>(
      x2h, 128, wkv, bkv, 128, 0, nullptr, nullptr, nullptr, kvh, 256, nullptr);
  // 4) attention (1024 independent 256x256 problems)
  attn_kernel<<<B_DIM * NHEAD * H_DIM, blk, 0, stream>>>(qh, kvh, oat);
  // 5) cat_mid = cat1 + o @ wo^T + bo  (f32, channel-major, into d_out)
  gemm_kernel<<<dim3(2, P_TOT / 128), blk, 0, stream>>>(
      oat, 128, wo, bo, 128, 3, nullptr, feats1, feats2, nullptr, 0, outf);
  // 6) LN(fn) -> pixel-major f16
  ln_mid_kernel<<<P_TOT / 256, blk, 0, stream>>>(outf, fn_g, fn_b, xyh);
  // 7) y = SiLU(xy @ w_in^T + b_in)
  gemm_kernel<<<dim3(4, P_TOT / 128), blk, 0, stream>>>(
      xyh, 128, w_in, b_in, 128, 1, nullptr, nullptr, nullptr, yh, 256, nullptr);
  // 8) depthwise 3/5/7 (+bias) -> concatenated 768ch
  dw_kernel<<<dim3(P_TOT / 256, E_CH / 8), blk, 0, stream>>>(
      yh, w_dw3, b_dw3, w_dw5, b_dw5, w_dw7, b_dw7, dwc);
  // 9) z = y + SiLU(dwcat @ w_pw^T + b_pw)
  gemm_kernel<<<dim3(4, P_TOT / 128), blk, 0, stream>>>(
      dwc, 768, w_pw, b_pw, 768, 2, yh, nullptr, nullptr, zh, 256, nullptr);
  // 10) d_out += z @ w_out^T + b_out
  gemm_kernel<<<dim3(2, P_TOT / 128), blk, 0, stream>>>(
      zh, 256, w_out, b_out, 256, 4, nullptr, nullptr, nullptr, nullptr, 0, outf);
}
